// Memory_Network_13443247637030
// MI455X (gfx1250) — compile-verified, matrix operand order checked by reading, not run on hardware
//
#include <hip/hip_runtime.h>

#define BS    128
#define MEMN  262144
#define SDIM  512
#define CDIM  313
#define TOPK  256
#define NBINS 2048
#define ALPHA 0.1f
#define THRES 0.005f

typedef __attribute__((ext_vector_type(16))) __bf16 v16bf;
typedef __attribute__((ext_vector_type(8)))  float  v8f;
typedef __attribute__((ext_vector_type(4)))  float  v4f;
typedef __attribute__((ext_vector_type(4)))  unsigned int v4u;
typedef __attribute__((ext_vector_type(8)))  int  v8i;
typedef __attribute__((ext_vector_type(4)))  int  v4i;

#ifdef __has_builtin
#if __has_builtin(__builtin_amdgcn_tensor_load_to_lds)
#define USE_TDM 1
#endif
#endif

union V16 {
    uint4          q[2];
    unsigned short u[16];
    v16bf          v;
};

__device__ __forceinline__ unsigned orderKey(float f) {
    unsigned b = __float_as_uint(f);
    return (b & 0x80000000u) ? ~b : (b | 0x80000000u);
}
__device__ __forceinline__ float fromKey(unsigned k) {
    unsigned b = (k & 0x80000000u) ? (k & 0x7fffffffu) : ~k;
    return __uint_as_float(b);
}

// ---------------------------------------------------------------------------
// 1) q = normalize(query); store as bf16 (row-major) for WMMA A-operand
// ---------------------------------------------------------------------------
__global__ void k_normalize(const float* __restrict__ query,
                            unsigned short* __restrict__ qb) {
    int row = blockIdx.x;
    __shared__ float red[256];
    float ss = 0.f;
    for (int k = threadIdx.x; k < SDIM; k += 256) {
        float v = query[row * SDIM + k];
        ss += v * v;
    }
    red[threadIdx.x] = ss;
    __syncthreads();
    for (int s = 128; s > 0; s >>= 1) {
        if (threadIdx.x < (unsigned)s) red[threadIdx.x] += red[threadIdx.x + s];
        __syncthreads();
    }
    float rn = rsqrtf(red[0] + 1e-24f);
    for (int k = threadIdx.x; k < SDIM; k += 256) {
        float v = query[row * SDIM + k] * rn;
        __bf16 h = (__bf16)v;
        unsigned short us;
        __builtin_memcpy(&us, &h, 2);
        qb[row * SDIM + k] = us;
    }
}

// ---------------------------------------------------------------------------
// 2) scores[BS][MEMN] = q_bf16 @ key^T via v_wmma_f32_16x16x32_bf16.
//    Block: 256 threads = 8 waves. Whole q (128x512 bf16 = 128KB) staged in
//    LDS by the Tensor Data Mover (one tensor_load_to_lds from wave 0);
//    each wave owns one 16-slot column group and sweeps all 8 batch tiles per
//    key-fragment load. Keys are streamed NON-TEMPORAL so the 512MB stream
//    does not evict the L2-resident scores array.
// ---------------------------------------------------------------------------
__global__ void k_gemm(const float* __restrict__ skey,
                       const unsigned short* __restrict__ qb,
                       float* __restrict__ scores) {
    extern __shared__ unsigned char smem_raw[];
    unsigned short* sq = (unsigned short*)smem_raw;

#ifdef USE_TDM
    if (threadIdx.x < 32) {
        // D# group0: count=1, lds_addr=0, global_addr=qb, type=2 ("image")
        unsigned long long ga = (unsigned long long)(uintptr_t)qb;
        v4u g0 = { 1u, 0u, (unsigned)(ga & 0xFFFFFFFFu),
                   (unsigned)(((ga >> 32) & 0x01FFFFFFu) | (2u << 30)) };
        // D# group1: data_size=2B(code 1); tensor 512x128, tile 512x128,
        // dim0_stride=512 (element units); no multicast/pad/iterate.
        v8i g1 = { (int)(1u << 16),      // wg_mask=0, data_size=1 (2 bytes)
                   (int)(512u << 16),    // tensor_dim0 = 512 (low16)
                   (int)(128u << 16),    // dim0 hi=0 | tensor_dim1 = 128 (low16)
                   (int)(512u << 16),    // dim1 hi=0 | tile_dim0 = 512
                   128,                  // tile_dim1 = 128, tile_dim2 = 0
                   512,                  // tensor_dim0_stride lo32 = 512
                   0, 0 };
        v4i gz4 = { 0, 0, 0, 0 };
        v8i gz8 = { 0, 0, 0, 0, 0, 0, 0, 0 };
        __builtin_amdgcn_tensor_load_to_lds(g0, g1, gz4, gz4, gz8, 0);
        __builtin_amdgcn_s_wait_tensorcnt(0);
    }
    __syncthreads();
#else
    {
        uint4*       sq4 = (uint4*)smem_raw;
        const uint4* gq4 = (const uint4*)qb;
        for (int i = threadIdx.x; i < (BS * SDIM) / 8; i += blockDim.x)
            sq4[i] = gq4[i];
        __syncthreads();
    }
#endif

    const int lane   = threadIdx.x & 31;
    const int wave   = threadIdx.x >> 5;
    const int n0     = blockIdx.x * 128 + wave * 16;
    const int nlane  = n0 + (lane & 15);
    const int kshift = (lane & 16) ? 8 : 0;   // K-chunk phase per lane half
    const float* krow = skey + (size_t)nlane * SDIM;

    v8f acc[8] = {};

    for (int ks = 0; ks < SDIM / 32; ++ks) {
        const int kb = ks * 32 + kshift;
        // prefetch next K-chunk of this key row
        __builtin_prefetch(krow + kb + 32, 0, 0);
        // B fragment: this lane's key row, K = {kb..kb+7, kb+16..kb+23},
        // fp32 -> bf16, loaded with non-temporal hint (single-use stream)
        v4f f0 = __builtin_nontemporal_load((const v4f*)(krow + kb));
        v4f f1 = __builtin_nontemporal_load((const v4f*)(krow + kb + 4));
        v4f f2 = __builtin_nontemporal_load((const v4f*)(krow + kb + 16));
        v4f f3 = __builtin_nontemporal_load((const v4f*)(krow + kb + 20));
        float bf[16] = {f0.x, f0.y, f0.z, f0.w, f1.x, f1.y, f1.z, f1.w,
                        f2.x, f2.y, f2.z, f2.w, f3.x, f3.y, f3.z, f3.w};
        V16 vb;
#pragma unroll
        for (int i = 0; i < 16; ++i) vb.v[i] = (__bf16)bf[i];

#pragma unroll
        for (int m = 0; m < 8; ++m) {
            const int arow = m * 16 + (lane & 15);
            V16 va;
            va.q[0] = *(const uint4*)(sq + arow * SDIM + kb);
            va.q[1] = *(const uint4*)(sq + arow * SDIM + kb + 16);
            acc[m] = __builtin_amdgcn_wmma_f32_16x16x32_bf16(
                false, va.v, false, vb.v, (short)0, acc[m], false, false);
        }
    }

    // C layout: VGPR r, lanes 0-15 -> M=r, lanes 16-31 -> M=8+r; N = lane%16
    const int col   = n0 + (lane & 15);
    const int rbase = (lane >> 4) * 8;
#pragma unroll
    for (int m = 0; m < 8; ++m)
#pragma unroll
        for (int r = 0; r < 8; ++r)
            scores[(size_t)(m * 16 + rbase + r) * MEMN + col] = acc[m][r];
}

// ---------------------------------------------------------------------------
// 3) per-row histogram of scores over [-1,1], LDS-privatized.
//    Each block: 256 threads x 32 scores = 8192 contiguous scores (one row).
//    Scores were just written -> should hit in L2 (they fit in 192MB).
// ---------------------------------------------------------------------------
__global__ void k_hist(const float* __restrict__ scores,
                       unsigned* __restrict__ hist) {
    __shared__ unsigned hloc[NBINS];
    for (int i = threadIdx.x; i < NBINS; i += 256) hloc[i] = 0u;
    __syncthreads();

    const int row = blockIdx.x >> 5;               // MEMN/8192 = 32 blocks/row
    const size_t base = (size_t)blockIdx.x * 8192;
    for (int i = threadIdx.x; i < 8192; i += 256) {
        float s = scores[base + i];
        int bin = (int)((s + 1.0f) * (NBINS * 0.5f));
        bin = min(max(bin, 0), NBINS - 1);
        atomicAdd(&hloc[bin], 1u);
    }
    __syncthreads();
    for (int i = threadIdx.x; i < NBINS; i += 256)
        if (hloc[i]) atomicAdd(&hist[row * NBINS + i], hloc[i]);
}

// ---------------------------------------------------------------------------
// 4) per-row top-k threshold tau (256th-largest, bin granularity);
//    also initialize pos/neg max keys to orderKey(0.0f).
// ---------------------------------------------------------------------------
__global__ void k_thresh(const unsigned* __restrict__ hist,
                         float* __restrict__ tau,
                         unsigned* __restrict__ pn) {
    int r = threadIdx.x;
    if (r < BS) {
        unsigned cnt = 0;
        float t = -1.0f;
        for (int b = NBINS - 1; b >= 0; --b) {
            cnt += hist[r * NBINS + b];
            if (cnt >= TOPK) { t = -1.0f + b * (2.0f / NBINS); break; }
        }
        tau[r] = t;
        pn[2 * r]     = orderKey(0.0f);  // pos (mask zeros participate in max)
        pn[2 * r + 1] = orderKey(0.0f);  // neg
    }
}

// ---------------------------------------------------------------------------
// 5) select score >= tau[row], compute KL(color_value[slot] || color_feat[row]),
//    classify, atomic-max into pos/neg via order-preserving uint key.
// ---------------------------------------------------------------------------
__global__ void k_select(const float* __restrict__ scores,
                         const float* __restrict__ tau,
                         const float* __restrict__ cval,
                         const float* __restrict__ cfeat,
                         unsigned* __restrict__ pn) {
    size_t idx = (size_t)blockIdx.x * blockDim.x + threadIdx.x;
    int row  = (int)(idx >> 18);           // MEMN = 2^18
    int slot = (int)(idx & (MEMN - 1));
    float s = scores[idx];
    if (s < tau[row]) return;

    const float* a = cval + (size_t)slot * CDIM;
    const float* b = cfeat + (size_t)row * CDIM;
    float kl = 0.f;
    for (int c = 0; c < CDIM; ++c) {
        float av = __builtin_nontemporal_load(a + c);   // one-shot gather
        kl += av * __log10f(av / (b[c] + 1e-8f));
    }
    int cls = (kl < THRES) ? 0 : 1;        // 0: pos (similar color), 1: neg
    atomicMax(&pn[row * 2 + cls], orderKey(s));
}

// ---------------------------------------------------------------------------
// 6) loss = mean(max(neg - pos + alpha, 0)) over rows — one wave32
// ---------------------------------------------------------------------------
__global__ void k_final(const unsigned* __restrict__ pn,
                        float* __restrict__ out) {
    int lane = threadIdx.x;
    float sum = 0.f;
    for (int r = lane; r < BS; r += 32) {
        float pos = fromKey(pn[2 * r]);
        float neg = fromKey(pn[2 * r + 1]);
        float h = neg - pos + ALPHA;
        sum += (h > 0.f) ? h : 0.f;
    }
    for (int off = 16; off; off >>= 1) sum += __shfl_xor(sum, off, 32);
    if (lane == 0) out[0] = sum * (1.0f / BS);
}

// ---------------------------------------------------------------------------
extern "C" void kernel_launch(void* const* d_in, const int* in_sizes, int n_in,
                              void* d_out, int out_size, void* d_ws, size_t ws_size,
                              hipStream_t stream) {
    const float* query    = (const float*)d_in[0];   // [BS, SDIM]
    const float* cfeat    = (const float*)d_in[1];   // [BS, CDIM]
    const float* skey     = (const float*)d_in[2];   // [MEMN, SDIM]
    const float* cval     = (const float*)d_in[3];   // [MEMN, CDIM]
    float* out = (float*)d_out;

    unsigned char* ws = (unsigned char*)d_ws;
    unsigned short* qb  = (unsigned short*)ws;                      // 128 KB
    float* scores = (float*)(ws + (size_t)(1 << 17));               // 134 MB
    unsigned* hist = (unsigned*)(ws + (size_t)(1 << 17) +
                                 (size_t)BS * MEMN * sizeof(float));
    float* tau   = (float*)((unsigned char*)hist + (size_t)BS * NBINS * 4);
    unsigned* pn = (unsigned*)((unsigned char*)tau + (size_t)BS * 4);

    (void)in_sizes; (void)n_in; (void)out_size; (void)ws_size;

    (void)hipMemsetAsync(hist, 0, (size_t)BS * NBINS * sizeof(unsigned), stream);

    k_normalize<<<BS, 256, 0, stream>>>(query, qb);

    // 2048 blocks x 256 threads; 128 KB dynamic LDS for the staged q matrix
    k_gemm<<<MEMN / 128, 256, BS * SDIM * 2, stream>>>(skey, qb, scores);

    k_hist<<<(BS * MEMN) / 8192, 256, 0, stream>>>(scores, hist);

    k_thresh<<<1, 128, 0, stream>>>(hist, tau, pn);

    k_select<<<(int)(((size_t)BS * MEMN) / 256), 256, 0, stream>>>(
        scores, tau, cval, cfeat, pn);

    k_final<<<1, 32, 0, stream>>>(pn, out);
}
